// DeepBRNNModel_3075196584538
// MI455X (gfx1250) — matrix operand order repaired and checked by reading, not compiled, not run on hardware
//
#include <hip/hip_runtime.h>
#include <cstddef>

// ---------------------------------------------------------------------------
// Problem constants (match reference)
// ---------------------------------------------------------------------------
#define VOC 50000
#define EMB 300
#define BB  64
#define TT  512
#define U1c 64
#define U2c 32
#define DDc 32

#define KPAD1 320          // E=300 padded to 10 chunks of K=32
#define N1    512          // 4*U1 * 2 directions (fwd cols 0..255, bwd 256..511)
#define N2    256          // 4*U2 * 2 directions
#define K2    128          // 2*U1 layer-2 input width (4 chunks of 32)

#define USE_TDM 1

typedef __attribute__((ext_vector_type(16))) __bf16 v16bf;
typedef __attribute__((ext_vector_type(8)))  float  v8f;
typedef __attribute__((ext_vector_type(4)))  unsigned int u32x4;
typedef __attribute__((ext_vector_type(8)))  int  i32x8;
typedef __attribute__((ext_vector_type(4)))  int  i32x4;

union BF16x16 { v16bf v; unsigned short u[16]; uint4 q[2]; };

__device__ inline unsigned short f2bf(float f) {
  union { __bf16 h; unsigned short u; } cv;
  cv.h = (__bf16)f;                 // native cvt, RTNE
  return cv.u;
}
__device__ inline float sigmoidf(float z) { return 1.0f / (1.0f + __expf(-z)); }

// A fragment: 16x32 bf16, row-major source. `row` already points at
// (laneRow * stride + kc*32 + hi*16). Elements 0..15 = K hi*16..+15 (contiguous).
__device__ inline v16bf load_a_frag(const unsigned short* row) {
  BF16x16 f;
  f.q[0] = *(const uint4*)(row);
  f.q[1] = *(const uint4*)(row + 8);
  return f.v;
}
// B fragment: 32x16 bf16 from W stored transposed (N,K) row-major.
// `row` points at (n * K + kc*32 + hi*16). Lane holds contiguous K range of 16.
__device__ inline v16bf load_b_frag(const unsigned short* row) {
  BF16x16 f;
  f.q[0] = *(const uint4*)(row);
  f.q[1] = *(const uint4*)(row + 8);
  return f.v;
}

// ---------------------------------------------------------------------------
// Weight prep: transpose to (N,K) bf16 with K padding, concat fwd|bwd, biases.
// ---------------------------------------------------------------------------
__global__ void prep_weights(const float* __restrict__ w1f, const float* __restrict__ b1f,
                             const float* __restrict__ w1b, const float* __restrict__ b1b,
                             const float* __restrict__ w2f, const float* __restrict__ b2f,
                             const float* __restrict__ w2b, const float* __restrict__ b2b,
                             unsigned short* __restrict__ W1xt,  // (512, 320)
                             unsigned short* __restrict__ W1ht,  // (512, 64)
                             unsigned short* __restrict__ W2xt,  // (256, 128)
                             unsigned short* __restrict__ W2ht,  // (256, 32)
                             float* __restrict__ bias1,          // 512
                             float* __restrict__ bias2)          // 256
{
  int i = blockIdx.x * blockDim.x + threadIdx.x;
  if (i < N1 * KPAD1) {               // W1xt[n][k] = w1{f,b}[k][n&255], k<300 else 0
    int n = i / KPAD1, k = i % KPAD1;
    const float* w = (n < 256) ? w1f : w1b;
    W1xt[i] = (k < EMB) ? f2bf(w[(size_t)k * 256 + (n & 255)]) : (unsigned short)0;
  }
  if (i < N1 * U1c) {                 // W1ht[n][k] = w1{f,b}[300+k][n&255]
    int n = i / U1c, k = i % U1c;
    const float* w = (n < 256) ? w1f : w1b;
    W1ht[i] = f2bf(w[(size_t)(EMB + k) * 256 + (n & 255)]);
  }
  if (i < N2 * K2) {                  // W2xt[n][k] = w2{f,b}[k][n&127]
    int n = i / K2, k = i % K2;
    const float* w = (n < 128) ? w2f : w2b;
    W2xt[i] = f2bf(w[(size_t)k * 128 + (n & 127)]);
  }
  if (i < N2 * U2c) {                 // W2ht[n][k] = w2{f,b}[128+k][n&127]
    int n = i / U2c, k = i % U2c;
    const float* w = (n < 128) ? w2f : w2b;
    W2ht[i] = f2bf(w[(size_t)(K2 + k) * 128 + (n & 127)]);
  }
  if (i < N1) bias1[i] = (i < 256) ? b1f[i] : b1b[i - 256];
  if (i < N2) bias2[i] = (i < 128) ? b2f[i] : b2b[i - 128];
}

// ---------------------------------------------------------------------------
// Embedding gather + f32->bf16 convert, done ONCE per element.
// Xbf[(t*64+b)][k] = bf16(emb[tok[b][t]][k]), K-padded with zeros to 320.
// ---------------------------------------------------------------------------
__global__ __launch_bounds__(256)
void gather_convert(const int* __restrict__ tok, const float* __restrict__ emb,
                    unsigned short* __restrict__ Xbf)
{
  const int groups = KPAD1 / 4;                       // 80 float4 groups per row
  const int g   = blockIdx.x * blockDim.x + threadIdx.x;
  const int row = g / groups;
  const int k0  = (g % groups) * 4;
  const int t = row >> 6, b = row & 63;
  ushort4 o;
  if (k0 < EMB) {                                     // EMB % 4 == 0 -> full vec ok
    const float4 x = *(const float4*)(emb + (size_t)tok[b * TT + t] * EMB + k0);
    o.x = f2bf(x.x); o.y = f2bf(x.y); o.z = f2bf(x.z); o.w = f2bf(x.w);
  } else {
    o.x = o.y = o.z = o.w = 0;
  }
  *(ushort4*)(Xbf + (size_t)row * KPAD1 + k0) = o;
}

// ---------------------------------------------------------------------------
// Unified bf16 GEMM: Z(M x N) = A(M x K bf16) @ Wt(N x K bf16)^T + bias.
// grid (M/64, N/128), block 256 (8 waves). Block's 64xK A-strip (contiguous
// in memory) is staged to LDS once via the Tensor Data Mover (1-D D#),
// then every wave streams A from LDS; B fragments are double-buffered.
// ---------------------------------------------------------------------------
template <int K, int N>
__global__ __launch_bounds__(256)
void gemm_bf16(const unsigned short* __restrict__ A,
               const unsigned short* __restrict__ Wt,
               const float* __restrict__ bias, float* __restrict__ Z)
{
  constexpr int KC = K / 32;
  __shared__ __align__(16) unsigned short As[64 * K];   // only LDS object -> offset 0
  const unsigned short* atile = A + (size_t)blockIdx.x * 64 * K;

#if USE_TDM && __has_builtin(__builtin_amdgcn_tensor_load_to_lds)
  if (threadIdx.x < 32) {                 // one wave drives the TDM
    constexpr unsigned NDW = (unsigned)(64 * K / 2);    // tile length in dwords
    const unsigned long long ga = (unsigned long long)atile;
    u32x4 g0;
    g0[0] = 1u;                                          // count=1, user descriptor
    g0[1] = 0u;                                          // lds_addr = 0 (As)
    g0[2] = (unsigned)ga;                                // global_addr[31:0]
    g0[3] = (unsigned)((ga >> 32) & 0x01FFFFFFu) | (2u << 30);  // addr[56:32] | type=2
    i32x8 g1;
    g1[0] = (int)(2u << 16);                 // data_size = 4 bytes; no mask/flags
    g1[1] = (int)((NDW & 0xFFFFu) << 16);    // tensor_dim0[15:0] @ bit 48
    g1[2] = (int)((NDW >> 16) | (1u << 16)); // tensor_dim0[31:16] | tensor_dim1 = 1
    g1[3] = (int)(NDW << 16);                // tensor_dim1 hi = 0 | tile_dim0 = NDW
    g1[4] = 1;                               // tile_dim1 = 1, tile_dim2 = 0
    g1[5] = (int)NDW;                        // tensor_dim0_stride (don't care, 1 row)
    g1[6] = 0;
    g1[7] = 0;
    i32x4 gz4;
    gz4[0] = gz4[1] = gz4[2] = gz4[3] = 0;   // groups 2/3 unused (<=2D tile)
    i32x8 gz8;
    gz8[0] = gz8[1] = gz8[2] = gz8[3] = 0;
    gz8[4] = gz8[5] = gz8[6] = gz8[7] = 0;
    __builtin_amdgcn_tensor_load_to_lds(g0, g1, gz4, gz4, gz8, 0);
    __builtin_amdgcn_s_wait_tensorcnt(0);
  }
  __syncthreads();
#else
  for (int i = threadIdx.x; i < 64 * K / 8; i += 256)
    ((uint4*)As)[i] = ((const uint4*)atile)[i];
  __syncthreads();
#endif

  const int wave = threadIdx.x >> 5, lane = threadIdx.x & 31;
  const int r16 = lane & 15, hi = lane >> 4;
  const int mb = blockIdx.x * 64;
  const int n  = blockIdx.y * 128 + wave * 16 + r16;
  const unsigned short* wrow = Wt + (size_t)n * K;

  v8f acc[4];
  const float bv = bias[n];
#pragma unroll
  for (int mt = 0; mt < 4; ++mt)
#pragma unroll
    for (int r = 0; r < 8; ++r) acc[mt][r] = bv;

  v16bf bf = load_b_frag(wrow + hi * 16);      // kc = 0
#pragma unroll
  for (int kc = 0; kc < KC; ++kc) {
    v16bf bnext{};
    if (kc + 1 < KC) bnext = load_b_frag(wrow + (kc + 1) * 32 + hi * 16);
#pragma unroll
    for (int mt = 0; mt < 4; ++mt) {
      const v16bf a = load_a_frag(As + (mt * 16 + r16) * K + kc * 32 + hi * 16);
      acc[mt] = __builtin_amdgcn_wmma_f32_16x16x32_bf16(false, a, false, bf,
                                                        (short)0, acc[mt], false, false);
    }
    bf = bnext;
  }
#pragma unroll
  for (int mt = 0; mt < 4; ++mt)
#pragma unroll
    for (int r = 0; r < 8; ++r) {
      const int m = mb + mt * 16 + r + 8 * hi;
      Z[(size_t)m * N + n] = acc[mt][r];
    }
}

// ---------------------------------------------------------------------------
// Persistent recurrent LSTM. One block per direction (blockIdx.x: 0=fwd,1=bwd).
// Per step: gates(64 x 4U) = Z[t] + H(64 x U) @ Wh(U x 4U) via WMMA,
// activations staged through LDS as f16, c kept in registers.
// Wh B-fragments live in registers for the whole 512-step loop.
// ---------------------------------------------------------------------------
template <int U, bool SEQ>
__global__ __launch_bounds__(8 * U)
void lstm_recur(const float* __restrict__ Z,          // (T*64, 8U) f32, dir cols dir*4U..
                const unsigned short* __restrict__ WhtAll, // (2*4U, U) bf16 transposed
                unsigned short* __restrict__ Yseq,    // (T*64, 2U) bf16, SEQ only
                float* __restrict__ hfinal)           // (64, 2U) f32, !SEQ only
{
  constexpr int NG = 4 * U;       // gate width per direction
  constexpr int ZS = 8 * U;       // Z row stride (both directions)
  constexpr int KC = U / 32;      // K chunks for the h-projection

  const int dir  = blockIdx.x;
  const int lane = threadIdx.x & 31;
  const int wave = threadIdx.x >> 5;      // 0 .. NG/16-1
  const int r16  = lane & 15, hi = lane >> 4;
  const int nb   = wave * 16;
  const int n    = nb + r16;              // gate column within this direction
  const int gate = nb / U;                // 0=f 1=i 2=g 3=o
  const int zcol = dir * NG + n;

  __shared__ __align__(16) unsigned short Hs[64 * U];  // h, bf16 row-major
  __shared__ __align__(16) _Float16      G[64 * NG];   // activated gates

  // zero initial hidden state
#pragma unroll
  for (int j = 0; j < 8; ++j) Hs[threadIdx.x + j * 8 * U] = 0;

  // resident B fragments of Wh for this wave's 16 gate columns
  const unsigned short* wrow = WhtAll + (size_t)(dir * NG + n) * U;
  v16bf bw[KC];
#pragma unroll
  for (int kc = 0; kc < KC; ++kc) bw[kc] = load_b_frag(wrow + kc * 32 + hi * 16);

  float c[8];
#pragma unroll
  for (int j = 0; j < 8; ++j) c[j] = 0.f;

  __syncthreads();

  for (int s = 0; s < TT; ++s) {
    const int t = dir ? (TT - 1 - s) : s;
    const float* zbase = Z + (size_t)t * BB * ZS + zcol;

    v8f acc[4];
#pragma unroll
    for (int mt = 0; mt < 4; ++mt)
#pragma unroll
      for (int r = 0; r < 8; ++r)
        acc[mt][r] = zbase[(size_t)(mt * 16 + r + 8 * hi) * ZS];

    if (s + 1 < TT) {  // hint next step's Z into cache
      const int tn = dir ? (TT - 2 - s) : (s + 1);
      __builtin_prefetch(Z + ((size_t)tn * BB + lane) * ZS + dir * NG, 0, 0);
    }

#pragma unroll
    for (int kc = 0; kc < KC; ++kc) {
#pragma unroll
      for (int mt = 0; mt < 4; ++mt) {
        v16bf a = load_a_frag(Hs + (mt * 16 + r16) * U + kc * 32 + hi * 16);
        acc[mt] = __builtin_amdgcn_wmma_f32_16x16x32_bf16(false, a, false, bw[kc],
                                                          (short)0, acc[mt], false, false);
      }
    }

    // activation + stage to LDS
#pragma unroll
    for (int mt = 0; mt < 4; ++mt)
#pragma unroll
      for (int r = 0; r < 8; ++r) {
        const float z = acc[mt][r];
        const float a = (gate == 2) ? tanhf(z) : sigmoidf(z);
        G[(mt * 16 + r + 8 * hi) * NG + n] = (_Float16)a;
      }
    __syncthreads();

    // elementwise cell update: 8 (m,u) elements per thread, c in registers
    const int base = threadIdx.x * 8;
#pragma unroll
    for (int j = 0; j < 8; ++j) {
      const int idx = base + j, m = idx / U, u = idx % U;
      const _Float16* gr = G + m * NG;
      const float f = (float)gr[u];
      const float i = (float)gr[U + u];
      const float g = (float)gr[2 * U + u];
      const float o = (float)gr[3 * U + u];
      const float cn = f * c[j] + i * g;
      c[j] = cn;
      const float h = o * tanhf(cn);
      const unsigned short hb = f2bf(h);
      Hs[m * U + u] = hb;
      if (SEQ) {
        Yseq[((size_t)t * BB + m) * (2 * U) + dir * U + u] = hb;
      } else if (s == TT - 1) {
        hfinal[m * (2 * U) + dir * U + u] = h;
      }
    }
    __syncthreads();
  }
}

// ---------------------------------------------------------------------------
// Dense head: out = sigmoid(relu(h2 @ wd + bd) @ wo + bo). 64 threads.
// ---------------------------------------------------------------------------
__global__ __launch_bounds__(64)
void head(const float* __restrict__ h2, const float* __restrict__ wd,
          const float* __restrict__ bd, const float* __restrict__ wo,
          const float* __restrict__ bo, float* __restrict__ out)
{
  const int b = threadIdx.x;
  const float* hr = h2 + b * (2 * U2c);
  float s2 = bo[0];
#pragma unroll 4
  for (int j = 0; j < DDc; ++j) {
    float s = bd[j];
    for (int k = 0; k < 2 * U2c; ++k) s += hr[k] * wd[k * DDc + j];
    s2 += fmaxf(s, 0.f) * wo[j];
  }
  out[b] = sigmoidf(s2);
}

// ---------------------------------------------------------------------------
extern "C" void kernel_launch(void* const* d_in, const int* in_sizes, int n_in,
                              void* d_out, int out_size, void* d_ws, size_t ws_size,
                              hipStream_t stream) {
  const int*   tok = (const int*)d_in[0];
  const float* emb = (const float*)d_in[1];
  const float* w1f = (const float*)d_in[2];  const float* b1f = (const float*)d_in[3];
  const float* w1b = (const float*)d_in[4];  const float* b1b = (const float*)d_in[5];
  const float* w2f = (const float*)d_in[6];  const float* b2f = (const float*)d_in[7];
  const float* w2b = (const float*)d_in[8];  const float* b2b = (const float*)d_in[9];
  const float* wd  = (const float*)d_in[10]; const float* bd  = (const float*)d_in[11];
  const float* wo  = (const float*)d_in[12]; const float* bo  = (const float*)d_in[13];
  float* outp = (float*)d_out;

  // workspace carve-up (256B aligned)
  size_t off = 0;
  auto carve = [&](size_t bytes) { size_t c = off; off += (bytes + 255) & ~(size_t)255; return c; };
  char* ws = (char*)d_ws;
  const size_t R = (size_t)TT * BB;                                   // 32768 rows
  float*          Z1    = (float*)(ws + carve(R * N1 * 4));           // 64 MB
  unsigned short* Xbf   = (unsigned short*)(ws + carve(R * KPAD1 * 2)); // 20 MB
  unsigned short* Y1    = (unsigned short*)(ws + carve(R * K2 * 2));  // 8 MB
  float*          Z2    = (float*)(ws + carve(R * N2 * 4));           // 32 MB
  unsigned short* W1xt  = (unsigned short*)(ws + carve((size_t)N1 * KPAD1 * 2));
  unsigned short* W1ht  = (unsigned short*)(ws + carve((size_t)N1 * U1c * 2));
  unsigned short* W2xt  = (unsigned short*)(ws + carve((size_t)N2 * K2 * 2));
  unsigned short* W2ht  = (unsigned short*)(ws + carve((size_t)N2 * U2c * 2));
  float*          bias1 = (float*)(ws + carve(N1 * 4));
  float*          bias2 = (float*)(ws + carve(N2 * 4));
  float*          h2    = (float*)(ws + carve((size_t)BB * 2 * U2c * 4));
  if (off > ws_size) return;  // insufficient scratch

  // 1) weight transpose/convert + bias concat
  prep_weights<<<dim3((N1 * KPAD1 + 255) / 256), 256, 0, stream>>>(
      w1f, b1f, w1b, b1b, w2f, b2f, w2b, b2b,
      W1xt, W1ht, W2xt, W2ht, bias1, bias2);
  // 2) embedding gather -> bf16 (once per element)
  gather_convert<<<dim3((unsigned)(R * (KPAD1 / 4) / 256)), 256, 0, stream>>>(tok, emb, Xbf);
  // 3) layer-1 input projection, both directions
  gemm_bf16<KPAD1, N1><<<dim3(R / 64, N1 / 128), 256, 0, stream>>>(Xbf, W1xt, bias1, Z1);
  // 4) layer-1 recurrence, fwd+bwd blocks, writes full sequence Y1
  lstm_recur<U1c, true><<<dim3(2), 8 * U1c, 0, stream>>>(Z1, W1ht, Y1, nullptr);
  // 5) layer-2 input projection
  gemm_bf16<K2, N2><<<dim3(R / 64, N2 / 128), 256, 0, stream>>>(Y1, W2xt, bias2, Z2);
  // 6) layer-2 recurrence, final hidden states only
  lstm_recur<U2c, false><<<dim3(2), 8 * U2c, 0, stream>>>(Z2, W2ht, nullptr, h2);
  // 7) dense head -> (64,1) probabilities
  head<<<dim3(1), 64, 0, stream>>>(h2, wd, bd, wo, bo, outp);
}
